// Model_34772055229103
// MI455X (gfx1250) — compile-verified, matrix operand order checked by reading, not run on hardware
//
#include <hip/hip_runtime.h>
#include <stdint.h>

#define D_STATE 8
#define D_FF    256
#define NB      32
#define NSCALES 4
#define ELAYERS 4

typedef __attribute__((ext_vector_type(16))) __bf16 v16bf;
typedef __attribute__((ext_vector_type(8)))  float  v8f;

enum { ACT_NONE=0, ACT_SILU=1, ACT_GELU=2, ACT_SOFTPLUS=3 };

static inline int ceildiv(int a, int b) { return (a + b - 1) / b; }

// ---------------------------------------------------------------------------
// Device helpers
// ---------------------------------------------------------------------------
__device__ __forceinline__ float act_apply(float v, int act) {
  switch (act) {
    case ACT_SILU:     return v / (1.f + __expf(-v));
    case ACT_GELU:     return 0.5f * v * (1.f + erff(v * 0.70710678118654752f));
    case ACT_SOFTPLUS: return v > 20.f ? v : log1pf(__expf(v));
    default:           return v;
  }
}

__device__ __forceinline__ void pack8(const float4& p, const float4& q,
                                      v16bf& d, int base) {
  d[base + 0] = (__bf16)p.x; d[base + 1] = (__bf16)p.y;
  d[base + 2] = (__bf16)p.z; d[base + 3] = (__bf16)p.w;
  d[base + 4] = (__bf16)q.x; d[base + 5] = (__bf16)q.y;
  d[base + 6] = (__bf16)q.z; d[base + 7] = (__bf16)q.w;
}

// ---------------------------------------------------------------------------
// FAST WMMA GEMM: requires K%32==0, N%32==0, M%16==0, lda%4==0.
// Y[M,N] = act( X[M,K](lda) @ W[N,K]^T + bias )  (+= if acc)
// One wave per 16x32 output: A operand shared by two WMMAs (2x A reuse,
// back-to-back independent matrix ops). All loads unconditional b128;
// no lane-dependent control flow in the K loop (EXEC stays all-ones).
// A-operand layout (ISA 7.12.2, 16-bit A 16x32): lane l holds row l&15;
// half h=l>>4 covers K chunks [h*8,h*8+8) and [16+h*8,16+h*8+8).
// ---------------------------------------------------------------------------
__global__ void k_gemm_wmma_fast(const float* __restrict__ X, int lda,
                                 const float* __restrict__ W,
                                 const float* __restrict__ bias,
                                 float* __restrict__ Y,
                                 int M, int N, int K, int act, int acc)
{
  const int lane = threadIdx.x & 31;
  const int r = lane & 15;
  const int h = lane >> 4;
  const int tn = blockIdx.x, tm = blockIdx.y;
  const int m  = tm * 16 + r;
  const int n0 = tn * 32 + r;
  const int n1 = n0 + 16;

  const float* xrow = X + (size_t)m  * lda;
  const float* wr0  = W + (size_t)n0 * K;
  const float* wr1  = W + (size_t)n1 * K;

  v8f c0, c1;
#pragma unroll
  for (int j = 0; j < 8; j++) { c0[j] = 0.f; c1[j] = 0.f; }
  if (acc) {
#pragma unroll
    for (int j = 0; j < 8; j++) {
      int mm = tm * 16 + j + 8 * h;
      c0[j] = Y[(size_t)mm * N + tn * 32 + r];
      c1[j] = Y[(size_t)mm * N + tn * 32 + 16 + r];
    }
  }

  for (int k0 = 0; k0 < K; k0 += 32) {
    const int kb = k0 + h * 8;
    float4 xa = *(const float4*)(xrow + kb);
    float4 xb = *(const float4*)(xrow + kb + 4);
    float4 xc = *(const float4*)(xrow + kb + 16);
    float4 xd = *(const float4*)(xrow + kb + 20);
    float4 p0 = *(const float4*)(wr0 + kb);
    float4 p1 = *(const float4*)(wr0 + kb + 4);
    float4 p2 = *(const float4*)(wr0 + kb + 16);
    float4 p3 = *(const float4*)(wr0 + kb + 20);
    float4 q0 = *(const float4*)(wr1 + kb);
    float4 q1 = *(const float4*)(wr1 + kb + 4);
    float4 q2 = *(const float4*)(wr1 + kb + 16);
    float4 q3 = *(const float4*)(wr1 + kb + 20);

    v16bf a, b0, b1;
    pack8(xa, xb, a, 0);  pack8(xc, xd, a, 8);
    pack8(p0, p1, b0, 0); pack8(p2, p3, b0, 8);
    pack8(q0, q1, b1, 0); pack8(q2, q3, b1, 8);

    c0 = __builtin_amdgcn_wmma_f32_16x16x32_bf16(false, a, false, b0,
                                                 (short)0, c0, false, false);
    c1 = __builtin_amdgcn_wmma_f32_16x16x32_bf16(false, a, false, b1,
                                                 (short)0, c1, false, false);
    if (k0 + 32 < K) {   // uniform scalar branch; streams next K chunk
      __builtin_prefetch(xrow + kb + 32, 0, 1);
      __builtin_prefetch(wr0 + kb + 32, 0, 1);
      __builtin_prefetch(wr1 + kb + 32, 0, 1);
    }
  }

#pragma unroll
  for (int j = 0; j < 8; j++) {
    int mm = tm * 16 + j + 8 * h;
    float v0 = c0[j], v1 = c1[j];
    if (bias) { v0 += bias[tn * 32 + r]; v1 += bias[tn * 32 + 16 + r]; }
    v0 = act_apply(v0, act);
    v1 = act_apply(v1, act);
    Y[(size_t)mm * N + tn * 32 + r]      = v0;
    Y[(size_t)mm * N + tn * 32 + 16 + r] = v1;
  }
}

// ---------------------------------------------------------------------------
// GENERAL WMMA GEMM for odd shapes (K=14, K=dtr<=16, N=18/20/24...).
// Branchless guarding: row pointers and K indices are CLAMPED so every
// load is unconditional and in-bounds; out-of-range values are zeroed
// with selects (v_cndmask), never with exec-mask branches.
// ---------------------------------------------------------------------------
__global__ void k_gemm_wmma_gen(const float* __restrict__ X, int lda,
                                const float* __restrict__ W,
                                const float* __restrict__ bias,
                                float* __restrict__ Y,
                                int M, int N, int K, int act, int acc)
{
  const int lane = threadIdx.x & 31;
  const int r = lane & 15;
  const int h = lane >> 4;
  const int tn = blockIdx.x, tm = blockIdx.y;
  const int m = tm * 16 + r;
  const int n = tn * 16 + r;
  const bool mvalid = (m < M);
  const bool nvalid = (n < N);
  const int mc = mvalid ? m : (M - 1);
  const int nc = nvalid ? n : (N - 1);

  const float* xrow = X + (size_t)mc * lda;
  const float* wrow = W + (size_t)nc * K;

  v8f c;
#pragma unroll
  for (int j = 0; j < 8; j++) c[j] = 0.f;
  if (acc) {
#pragma unroll
    for (int j = 0; j < 8; j++) {
      int mm = tm * 16 + j + 8 * h;
      int nn = tn * 16 + r;
      if (mm < M && nn < N) c[j] = Y[(size_t)mm * N + nn];
    }
  }

  for (int k0 = 0; k0 < K; k0 += 32) {
    v16bf a, b;
#pragma unroll
    for (int i = 0; i < 8; i++) {
      int k1 = k0 + h * 8 + i;
      int k2 = k0 + 16 + h * 8 + i;
      int k1c = (k1 < K) ? k1 : (K - 1);      // clamp: load stays in-bounds
      int k2c = (k2 < K) ? k2 : (K - 1);
      float fa1 = xrow[k1c];
      float fa2 = xrow[k2c];
      float fb1 = wrow[k1c];
      float fb2 = wrow[k2c];
      fa1 = (mvalid && k1 < K) ? fa1 : 0.f;   // select, not branch
      fa2 = (mvalid && k2 < K) ? fa2 : 0.f;
      fb1 = (nvalid && k1 < K) ? fb1 : 0.f;
      fb2 = (nvalid && k2 < K) ? fb2 : 0.f;
      a[i]     = (__bf16)fa1;
      a[8 + i] = (__bf16)fa2;
      b[i]     = (__bf16)fb1;
      b[8 + i] = (__bf16)fb2;
    }
    c = __builtin_amdgcn_wmma_f32_16x16x32_bf16(false, a, false, b,
                                                (short)0, c, false, false);
  }

#pragma unroll
  for (int j = 0; j < 8; j++) {
    int mm = tm * 16 + j + 8 * h;
    int nn = tn * 16 + r;
    if (mm < M && nn < N) {
      float v = c[j];
      if (bias) v += bias[nn];
      v = act_apply(v, act);
      Y[(size_t)mm * N + nn] = v;
    }
  }
}

// ---------------------------------------------------------------------------
// Depthwise causal conv (k=4) + bias + SiLU. Input = first half of xz
// (row stride 2*di). Output xc [B,L,di].
// ---------------------------------------------------------------------------
__global__ void k_conv_silu(const float* __restrict__ xz,
                            const float* __restrict__ w,
                            const float* __restrict__ cb,
                            float* __restrict__ xc,
                            int Bn, int L, int di)
{
  int idx = blockIdx.x * blockDim.x + threadIdx.x;
  int tot = Bn * L * di;
  if (idx >= tot) return;
  int d = idx % di;
  int t = idx / di;
  int l = t % L;
  int b = t / L;
  float acc = cb[d];
#pragma unroll
  for (int j = 0; j < 4; j++) {
    int ls = l + j - 3;
    if (ls >= 0)
      acc += xz[((size_t)(b * L + ls)) * (2 * di) + d] * w[d * 4 + j];
  }
  xc[(size_t)t * di + d] = acc / (1.f + __expf(-acc));
}

// ---------------------------------------------------------------------------
// Selective scan. One thread per (b,d) channel, serial over L.
// y = (scan + xc*D) * silu(z)
// ---------------------------------------------------------------------------
__global__ void k_scan(const float* __restrict__ dt,
                       const float* __restrict__ xdbc,
                       const float* __restrict__ xc,
                       const float* __restrict__ xz,
                       const float* __restrict__ A_log,
                       const float* __restrict__ Dp,
                       float* __restrict__ y,
                       int Bn, int L, int di, int R, int dtr)
{
  int tid = blockIdx.x * blockDim.x + threadIdx.x;
  if (tid >= Bn * di) return;
  int b = tid / di, d = tid - b * di;
  float A[D_STATE];
#pragma unroll
  for (int n = 0; n < D_STATE; n++) A[n] = -__expf(A_log[d * D_STATE + n]);
  float Dv = Dp[d];
  float h[D_STATE];
#pragma unroll
  for (int n = 0; n < D_STATE; n++) h[n] = 0.f;

  for (int l = 0; l < L; l++) {
    int t = b * L + l;
    float dtv = dt[(size_t)t * di + d];
    float xcv = xc[(size_t)t * di + d];
    const float* bc = xdbc + (size_t)t * R + dtr;   // Bm[0..7], Cm[0..7]
    float accv = 0.f;
#pragma unroll
    for (int n = 0; n < D_STATE; n++) {
      float dA = __expf(dtv * A[n]);
      h[n] = dA * h[n] + dtv * bc[n] * xcv;
      accv += h[n] * bc[D_STATE + n];
    }
    float zv = xz[(size_t)t * (2 * di) + di + d];
    float sz = zv / (1.f + __expf(-zv));
    y[(size_t)t * di + d] = (accv + xcv * Dv) * sz;
  }
}

// ---------------------------------------------------------------------------
// LayerNorm (optional residual add). Warp per token, shuffle reductions.
// ---------------------------------------------------------------------------
__global__ void k_ln(const float* __restrict__ x, const float* __restrict__ res,
                     const float* __restrict__ s, const float* __restrict__ b,
                     float* __restrict__ y, int T, int F)
{
  int lane = threadIdx.x & 31;
  int w = threadIdx.x >> 5;
  int t = blockIdx.x * 8 + w;
  if (t >= T) return;
  int C = F >> 5;
  float v[8];
  float sum = 0.f;
#pragma unroll
  for (int c = 0; c < 8; c++) {
    if (c < C) {
      int i = lane + c * 32;
      float vv = x[(size_t)t * F + i];
      if (res) vv += res[(size_t)t * F + i];
      v[c] = vv;
      sum += vv;
    }
  }
#pragma unroll
  for (int off = 16; off > 0; off >>= 1) sum += __shfl_xor(sum, off, 32);
  float mean = sum / (float)F;
  float var = 0.f;
#pragma unroll
  for (int c = 0; c < 8; c++) {
    if (c < C) { float d = v[c] - mean; var += d * d; }
  }
#pragma unroll
  for (int off = 16; off > 0; off >>= 1) var += __shfl_xor(var, off, 32);
  float rs = rsqrtf(var / (float)F + 1e-5f);
#pragma unroll
  for (int c = 0; c < 8; c++) {
    if (c < C) {
      int i = lane + c * 32;
      y[(size_t)t * F + i] = (v[c] - mean) * rs * s[i] + b[i];
    }
  }
}

// ---------------------------------------------------------------------------
// Sequence flip: y[b, L-1-l, f] = x[b, l, f]
// ---------------------------------------------------------------------------
__global__ void k_flip(const float* __restrict__ x, float* __restrict__ y,
                       int Bn, int L, int F)
{
  int idx = blockIdx.x * blockDim.x + threadIdx.x;
  int tot = Bn * L * F;
  if (idx >= tot) return;
  int f = idx % F;
  int t = idx / F;
  int l = t % L;
  int b = t / L;
  y[((size_t)(b * L + (L - 1 - l))) * F + f] = x[idx];
}

// ---------------------------------------------------------------------------
// Batched transpose: y[b,c,r] = x[b,r,c]
// ---------------------------------------------------------------------------
__global__ void k_transpose(const float* __restrict__ x, float* __restrict__ y,
                            int Bn, int R, int C)
{
  int idx = blockIdx.x * blockDim.x + threadIdx.x;
  int tot = Bn * R * C;
  if (idx >= tot) return;
  int c = idx % C;
  int t = idx / C;
  int r = t % R;
  int b = t / R;
  y[((size_t)b * C + c) * R + r] = x[idx];
}

// ---------------------------------------------------------------------------
// Per-batch dot with long weight vector (scale head): out[b*S+col]
// ---------------------------------------------------------------------------
__global__ void k_rowdot(const float* __restrict__ x, const float* __restrict__ w,
                         const float* __restrict__ bias, float* __restrict__ out,
                         int K, int S, int col)
{
  __shared__ float red[256];
  int b = blockIdx.x;
  int t = threadIdx.x;
  float s = 0.f;
  for (int k = t; k < K; k += 256) s += x[(size_t)b * K + k] * w[k];
  red[t] = s;
  __syncthreads();
  for (int o = 128; o > 0; o >>= 1) {
    if (t < o) red[t] += red[t + o];
    __syncthreads();
  }
  if (t == 0) out[b * S + col] = red[0] + bias[0];
}

// final: out[b] = sum_i outs[b,i]*fw[i] + fb
__global__ void k_final(const float* __restrict__ outs, const float* __restrict__ fw,
                        const float* __restrict__ fb, float* __restrict__ o,
                        int Bn, int S)
{
  int b = blockIdx.x * blockDim.x + threadIdx.x;
  if (b >= Bn) return;
  float s = fb[0];
  for (int i = 0; i < S; i++) s += outs[b * S + i] * fw[i];
  o[b] = s;
}

// ---------------------------------------------------------------------------
// Host-side parameter pytree parsing (setup_inputs dict order)
// ---------------------------------------------------------------------------
struct MambaP { const float *in_proj,*conv_w,*conv_b,*x_proj,*dt_w,*dt_b,*A_log,*D,*out_proj; };
struct LayerP { MambaP mf, mb; const float *c1w,*c1b,*c2w,*c2b,*n1s,*n1b,*n2s,*n2b; };
struct EncP   { LayerP ly[ELAYERS]; const float *ns,*nb; };
struct ScaleP { EncP time, space; const float *lin_w,*lin_b; };
struct MergeP { const float *w,*b; };

static MambaP rdM(void* const* in, int& p) {
  MambaP m;
  m.in_proj=(const float*)in[p++]; m.conv_w=(const float*)in[p++]; m.conv_b=(const float*)in[p++];
  m.x_proj=(const float*)in[p++]; m.dt_w=(const float*)in[p++]; m.dt_b=(const float*)in[p++];
  m.A_log=(const float*)in[p++]; m.D=(const float*)in[p++]; m.out_proj=(const float*)in[p++];
  return m;
}
static LayerP rdL(void* const* in, int& p) {
  LayerP l; l.mf = rdM(in,p); l.mb = rdM(in,p);
  l.c1w=(const float*)in[p++]; l.c1b=(const float*)in[p++];
  l.c2w=(const float*)in[p++]; l.c2b=(const float*)in[p++];
  l.n1s=(const float*)in[p++]; l.n1b=(const float*)in[p++];
  l.n2s=(const float*)in[p++]; l.n2b=(const float*)in[p++];
  return l;
}
static EncP rdE(void* const* in, int& p) {
  EncP e;
  for (int i = 0; i < ELAYERS; i++) e.ly[i] = rdL(in,p);
  e.ns=(const float*)in[p++]; e.nb=(const float*)in[p++];
  return e;
}

struct WS { float *xz,*xc,*dt,*yb,*xdbc,*bufA,*bufB,*newx,*x1,*xflip,*outs; };

static void gemm(hipStream_t st, const float* X, int lda, const float* W,
                 const float* bias, float* Y, int M, int N, int K, int act, int acc)
{
  if ((K & 31) == 0 && (N & 31) == 0 && (M & 15) == 0 && (lda & 3) == 0) {
    dim3 g(N / 32, M / 16);
    k_gemm_wmma_fast<<<g, 32, 0, st>>>(X, lda, W, bias, Y, M, N, K, act, acc);
  } else {
    dim3 g(ceildiv(N, 16), ceildiv(M, 16));
    k_gemm_wmma_gen<<<g, 32, 0, st>>>(X, lda, W, bias, Y, M, N, K, act, acc);
  }
}

static void mamba(hipStream_t st, WS& w, const MambaP& p, const float* X,
                  float* out, int acc, int flip_out, int L, int dm)
{
  const int Bn = NB, di = 4 * dm, dtr = (dm + 15) / 16, R = dtr + 2 * D_STATE;
  const int T = Bn * L;
  gemm(st, X, dm, p.in_proj, nullptr, w.xz, T, 2 * di, dm, ACT_NONE, 0);
  k_conv_silu<<<ceildiv(T * di, 256), 256, 0, st>>>(w.xz, p.conv_w, p.conv_b, w.xc, Bn, L, di);
  gemm(st, w.xc, di, p.x_proj, nullptr, w.xdbc, T, R, di, ACT_NONE, 0);
  gemm(st, w.xdbc, R, p.dt_w, p.dt_b, w.dt, T, di, dtr, ACT_SOFTPLUS, 0);
  k_scan<<<ceildiv(Bn * di, 256), 256, 0, st>>>(w.dt, w.xdbc, w.xc, w.xz,
                                                p.A_log, p.D, w.yb, Bn, L, di, R, dtr);
  const float* ys = w.yb;
  if (flip_out) {   // backward mamba: flip gated y before out_proj (xc is free now)
    k_flip<<<ceildiv(T * di, 256), 256, 0, st>>>(w.yb, w.xc, Bn, L, di);
    ys = w.xc;
  }
  gemm(st, ys, di, p.out_proj, nullptr, out, T, dm, di, ACT_NONE, acc);
}

static void encoder(hipStream_t st, WS& w, const EncP& e, float* cur, int L, int F)
{
  const int Bn = NB, T = Bn * L;
  for (int i = 0; i < ELAYERS; i++) {
    const LayerP& lp = e.ly[i];
    mamba(st, w, lp.mf, cur, w.newx, 0, 0, L, F);
    k_flip<<<ceildiv(T * F, 256), 256, 0, st>>>(cur, w.xflip, Bn, L, F);
    mamba(st, w, lp.mb, w.xflip, w.newx, 1, 1, L, F);
    k_ln<<<ceildiv(T, 8), 256, 0, st>>>(w.newx, nullptr, lp.n1s, lp.n1b, w.x1, T, F);
    gemm(st, w.x1, F, lp.c1w, lp.c1b, w.xz, T, D_FF, F, ACT_GELU, 0);    // h (alias xz)
    gemm(st, w.xz, D_FF, lp.c2w, lp.c2b, w.dt, T, F, D_FF, ACT_NONE, 0); // y2 (alias dt)
    k_ln<<<ceildiv(T, 8), 256, 0, st>>>(w.x1, w.dt, lp.n2s, lp.n2b, cur, T, F);
  }
  k_ln<<<ceildiv(T, 8), 256, 0, st>>>(cur, nullptr, e.ns, e.nb, cur, T, F);
}

// ---------------------------------------------------------------------------
extern "C" void kernel_launch(void* const* d_in, const int* in_sizes, int n_in,
                              void* d_out, int out_size, void* d_ws, size_t ws_size,
                              hipStream_t stream)
{
  (void)in_sizes; (void)n_in; (void)out_size; (void)ws_size;

  int p = 0;
  const float* x_enc     = (const float*)d_in[p++];
  const float* local_W   = (const float*)d_in[p++];
  const float* local_pos = (const float*)d_in[p++];
  const float* space_W   = (const float*)d_in[p++];
  const float* space_pos = (const float*)d_in[p++];
  ScaleP scales[NSCALES];
  for (int i = 0; i < NSCALES; i++) {
    scales[i].time  = rdE(d_in, p);
    scales[i].space = rdE(d_in, p);
    scales[i].lin_w = (const float*)d_in[p++];
    scales[i].lin_b = (const float*)d_in[p++];
  }
  MergeP merge[NSCALES - 1];
  for (int i = 0; i < NSCALES - 1; i++) {
    merge[i].w = (const float*)d_in[p++];
    merge[i].b = (const float*)d_in[p++];
  }
  const float* final_w = (const float*)d_in[p++];
  const float* final_b = (const float*)d_in[p++];

  // Workspace carving (floats). Maxima over pipeline: T*di <= 8192*256.
  float* base = (float*)d_ws;
  size_t off = 0;
  WS w;
  w.xz    = base + off; off += 4194304;  // T*2di max (also FFN hidden alias)
  w.xc    = base + off; off += 2097152;  // T*di max
  w.dt    = base + off; off += 2097152;  // T*di max (also FFN out alias)
  w.yb    = base + off; off += 2097152;  // T*di max
  w.xdbc  = base + off; off += 262144;   // T*R max
  w.bufA  = base + off; off += 524288;   // activations
  w.bufB  = base + off; off += 524288;
  w.newx  = base + off; off += 524288;
  w.x1    = base + off; off += 524288;
  w.xflip = base + off; off += 524288;
  w.outs  = base + off; off += 256;

  // x = x_enc.swapaxes(1,2): [32,32,14] -> [32,14,32]
  k_transpose<<<ceildiv(NB * 32 * 14, 256), 256, 0, stream>>>(x_enc, w.bufA, NB, 32, 14);
  // local patch embed: [448,32] @ [256,32]^T + pos -> [32,14,256]
  gemm(stream, w.bufA, 32, local_W, local_pos, w.bufB, NB * 14, 256, 32, ACT_NONE, 0);

  float* cur = w.bufB;
  float* alt = w.bufA;
  int L = 14, F = 256;

  for (int i = 0; i < NSCALES; i++) {
    // time encoder over [B, L, F] (dm = F)
    encoder(stream, w, scales[i].time, cur, L, F);
    // swap to [B, F, L]
    k_transpose<<<ceildiv(NB * L * F, 256), 256, 0, stream>>>(cur, alt, NB, L, F);
    { float* t = cur; cur = alt; alt = t; }
    { int t = L; L = F; F = t; }
    if (i == 0) {
      // space patch embed: K=14 -> 64 with pos bias
      gemm(stream, cur, 14, space_W, space_pos, alt, NB * L, 64, 14, ACT_NONE, 0);
      { float* t = cur; cur = alt; alt = t; }
      F = 64;
    }
    // space encoder over [B, L, 64]
    encoder(stream, w, scales[i].space, cur, L, F);
    // scale head: out[b, i] = x[b].flatten() . lin_w + lin_b
    k_rowdot<<<NB, 256, 0, stream>>>(cur, scales[i].lin_w, scales[i].lin_b,
                                     w.outs, L * F, NSCALES, i);
    if (i < NSCALES - 1) {
      // merge: [B,L,64] viewed [B,L/2,128] @ [64,128]^T + b -> [B,L/2,64]
      gemm(stream, cur, 2 * F, merge[i].w, merge[i].b, alt, NB * (L / 2), F, 2 * F,
           ACT_NONE, 0);
      // transpose -> [B,64,L/2]
      k_transpose<<<ceildiv(NB * (L / 2) * F, 256), 256, 0, stream>>>(alt, cur, NB, L / 2, F);
      F = L / 2;
      L = 64;
    }
  }

  k_final<<<1, 32, 0, stream>>>(w.outs, final_w, final_b, (float*)d_out, NB, NSCALES);
}
